// UNetTransformerBlock_54443005444484
// MI455X (gfx1250) — matrix operand order repaired and checked by reading, not compiled, not run on hardware
//
#include <hip/hip_runtime.h>
#include <hip/hip_bf16.h>

// ---------------------------------------------------------------------------
// Types / constants
// ---------------------------------------------------------------------------
typedef __attribute__((ext_vector_type(16))) _Float16 v16h;
typedef __attribute__((ext_vector_type(8)))  _Float16 v8h;
typedef __attribute__((ext_vector_type(8)))  float    v8f;

#define NB   2
#define HH   256
#define WW   256
#define PIX  (NB*HH*WW)          // 131072 pixels full-res
#define PIXD (NB*128*128)        // 32768 pixels quarter-res
#define DEC_OFF (NB*64*HH*WW)    // decoded output floats before temporal_out
#define EPSX 1e-5f

// ---------------------------------------------------------------------------
// WMMA helpers (CDNA5 wave32, v_wmma_f32_16x16x32_f16)
// ---------------------------------------------------------------------------
__device__ __forceinline__ v8f wmma16(v16h a, v16h b, v8f c) {
  return __builtin_amdgcn_wmma_f32_16x16x32_f16(false, a, false, b,
                                                (short)0, c, false, false);
}

// A fragment: 16x32 f16, row-major source M[ld] (ISA 7.12.2 A layout)
__device__ __forceinline__ v16h ld_a16(const _Float16* M, int ld, int row0,
                                       int kk, int lane) {
  int m  = row0 + (lane & 15);
  int kg = (lane >> 4) * 8;
  const _Float16* p = M + m * ld + kk + kg;
  v8h lo = *(const v8h*)p;
  v8h hi = *(const v8h*)(p + 16);
  v16h r;
#pragma unroll
  for (int i = 0; i < 8; ++i) { r[i] = lo[i]; r[i + 8] = hi[i]; }
  return r;
}

// B fragment: 32x16 f16 from row-major W[n][k] (lane = column, K contiguous)
__device__ __forceinline__ v16h ld_b16(const _Float16* M, int ld, int n0,
                                       int kk, int lane) {
  int nn = n0 + (lane & 15);
  const _Float16* p = M + nn * ld + kk + (lane >> 4) * 8;
  v8h lo = *(const v8h*)p;
  v8h hi = *(const v8h*)(p + 16);
  v16h r;
#pragma unroll
  for (int i = 0; i < 8; ++i) { r[i] = lo[i]; r[i + 8] = hi[i]; }
  return r;
}

__device__ __forceinline__ float redmax16(float v) {
  v = fmaxf(v, __shfl_xor(v, 1, 32));
  v = fmaxf(v, __shfl_xor(v, 2, 32));
  v = fmaxf(v, __shfl_xor(v, 4, 32));
  v = fmaxf(v, __shfl_xor(v, 8, 32));
  return v;
}
__device__ __forceinline__ float redsum16(float v) {
  v += __shfl_xor(v, 1, 32);
  v += __shfl_xor(v, 2, 32);
  v += __shfl_xor(v, 4, 32);
  v += __shfl_xor(v, 8, 32);
  return v;
}

// ---------------------------------------------------------------------------
// K0: convert weights to f16, pad output-channel dim to multiples of 16
// su_w 189x128 -> 192x128 ; ds_w 125x64 -> 128x64 ; w1 128x64 ; w2 61x128 -> 64x128
// ---------------------------------------------------------------------------
__global__ __launch_bounds__(256) void k_wconv(const float* su_w, const float* ds_w,
                                               const float* w1, const float* w2,
                                               _Float16* su_wh, _Float16* ds_wh,
                                               _Float16* w1h, _Float16* w2h) {
  int i = blockIdx.x * 256 + threadIdx.x;
  if (i < 192 * 128) {
    int o = i >> 7, c = i & 127;
    su_wh[i] = (_Float16)(o < 189 ? su_w[o * 128 + c] : 0.f);
    return;
  }
  i -= 192 * 128;
  if (i < 128 * 64) {
    int o = i >> 6, c = i & 63;
    ds_wh[i] = (_Float16)(o < 125 ? ds_w[o * 64 + c] : 0.f);
    return;
  }
  i -= 128 * 64;
  if (i < 128 * 64) { w1h[i] = (_Float16)w1[i]; return; }
  i -= 128 * 64;
  if (i < 64 * 128) {
    int o = i >> 7, c = i & 127;
    w2h[i] = (_Float16)(o < 61 ? w2[o * 128 + c] : 0.f);
  }
}

// ---------------------------------------------------------------------------
// K1: bilinear-upsample ds_latent, concat with skip_latent, image layernorm,
// emit f16 GEMM operand X[pix][128]; also drop radiance into kv_img[125..127]
// ---------------------------------------------------------------------------
__global__ __launch_bounds__(256) void k_prep_su(const float* skip, const float* dsl,
                                                 const float* rad, const float* g,
                                                 const float* b, _Float16* X,
                                                 _Float16* kv_img) {
  int p = blockIdx.x * 256 + threadIdx.x;
  int n = p >> 16, rem = p & 65535, y = rem >> 8, x = rem & 255;

  float sy = (y + 0.5f) * 0.5f - 0.5f, fy = floorf(sy), ty = sy - fy;
  float sx = (x + 0.5f) * 0.5f - 0.5f, fx = floorf(sx), tx = sx - fx;
  int y0 = min(max((int)fy, 0), 127), y1 = min(max((int)fy + 1, 0), 127);
  int x0 = min(max((int)fx, 0), 127), x1 = min(max((int)fx + 1, 0), 127);

  float vals[128];
  const float* sp = skip + ((size_t)n * 64) * 65536 + rem;
#pragma unroll 8
  for (int c = 0; c < 64; ++c) vals[c] = sp[(size_t)c * 65536];
  const float* dp = dsl + ((size_t)n * 64) * 16384;
#pragma unroll 4
  for (int c = 0; c < 64; ++c) {
    const float* pl = dp + (size_t)c * 16384;
    float v00 = pl[y0 * 128 + x0], v01 = pl[y0 * 128 + x1];
    float v10 = pl[y1 * 128 + x0], v11 = pl[y1 * 128 + x1];
    float top = v00 + (v01 - v00) * tx, bot = v10 + (v11 - v10) * tx;
    vals[64 + c] = top + (bot - top) * ty;
  }
  float mu = 0.f;
#pragma unroll
  for (int c = 0; c < 128; ++c) mu += vals[c];
  mu *= (1.f / 128.f);
  float var = 0.f;
#pragma unroll
  for (int c = 0; c < 128; ++c) { float d = vals[c] - mu; var += d * d; }
  var *= (1.f / 128.f);
  float inv = rsqrtf(var + EPSX);
  _Float16* xr = X + (size_t)p * 128;
#pragma unroll
  for (int c = 0; c < 128; ++c)
    xr[c] = (_Float16)((vals[c] - mu) * inv * g[c] + b[c]);

  const float* rp = rad + ((size_t)n * 3) * 65536 + rem;
#pragma unroll
  for (int j = 0; j < 3; ++j)
    kv_img[(size_t)p * 128 + 125 + j] = (_Float16)rp[(size_t)j * 65536];
}

// ---------------------------------------------------------------------------
// K2: motion-vector warp of temporal_kv (grid == full res, so the trailing
// bilinear_resize in the reference is identity). Emits f16 kv_t[pix][128].
// ---------------------------------------------------------------------------
__global__ __launch_bounds__(256) void k_warp(const float* tkv, const float* mv,
                                              _Float16* kv_t) {
  int p = blockIdx.x * 256 + threadIdx.x;
  int n = p >> 16, rem = p & 65535, y = rem >> 8, x = rem & 255;
  float mv0 = mv[((size_t)n * 2 + 0) * 65536 + rem];
  float mv1 = mv[((size_t)n * 2 + 1) * 65536 + rem];
  float xs = -1.f + 2.f * (float)x / 255.f;
  float ys = -1.f + 2.f * (float)y / 255.f;
  float gx = mv0 * (2.f / 256.f) + xs;
  float gy = ys - mv1 * (2.f / 256.f);
  float ix = fminf(fmaxf(((gx + 1.f) * 256.f - 1.f) * 0.5f, 0.f), 255.f);
  float iy = fminf(fmaxf(((gy + 1.f) * 256.f - 1.f) * 0.5f, 0.f), 255.f);
  float fx = floorf(ix), fy = floorf(iy);
  float tx = ix - fx, ty = iy - fy;
  int x0 = (int)fx, y0 = (int)fy;
  int x1 = min(x0 + 1, 255), y1 = min(y0 + 1, 255);
  const float* base = tkv + (size_t)n * 128 * 65536;
  _Float16* dst = kv_t + (size_t)p * 128;
#pragma unroll 4
  for (int c = 0; c < 128; ++c) {
    const float* pl = base + (size_t)c * 65536;
    float v00 = pl[y0 * 256 + x0], v01 = pl[y0 * 256 + x1];
    float v10 = pl[y1 * 256 + x0], v11 = pl[y1 * 256 + x1];
    float top = v00 + (v01 - v00) * tx, bot = v10 + (v11 - v10) * tx;
    dst[c] = (_Float16)(top + (bot - top) * ty);
  }
}

// ---------------------------------------------------------------------------
// K3: quarter-res layernorm of ds_latent -> f16 Xds; also seed ds_img[125..127]
// ---------------------------------------------------------------------------
__global__ __launch_bounds__(256) void k_ds_ln(const float* dsl, const float* g,
                                               const float* b, _Float16* Xds,
                                               _Float16* ds_img) {
  int p = blockIdx.x * 256 + threadIdx.x;   // quarter-res pixel
  int n = p >> 14, rem = p & 16383;
  float vals[64];
  const float* base = dsl + ((size_t)n * 64) * 16384 + rem;
#pragma unroll 8
  for (int c = 0; c < 64; ++c) vals[c] = base[(size_t)c * 16384];
  float mu = 0.f;
#pragma unroll
  for (int c = 0; c < 64; ++c) mu += vals[c];
  mu *= (1.f / 64.f);
  float var = 0.f;
#pragma unroll
  for (int c = 0; c < 64; ++c) { float d = vals[c] - mu; var += d * d; }
  var *= (1.f / 64.f);
  float inv = rsqrtf(var + EPSX);
  _Float16* xr = Xds + (size_t)p * 64;
#pragma unroll
  for (int c = 0; c < 64; ++c)
    xr[c] = (_Float16)((vals[c] - mu) * inv * g[c] + b[c]);
#pragma unroll
  for (int j = 0; j < 3; ++j)
    ds_img[(size_t)p * 128 + 125 + j] = (_Float16)vals[61 + j];
}

// ---------------------------------------------------------------------------
// K4: ds-branch conv1x1 GEMM via WMMA: ds_img[:,0:125] = Xds @ ds_w^T
// one 16x16 output tile per wave; K=64 (2 wmma k-steps)
// ---------------------------------------------------------------------------
__global__ __launch_bounds__(256) void k_gemm_ds(const _Float16* Xds,
                                                 const _Float16* Wh,
                                                 _Float16* ds_img) {
  int wave = (blockIdx.x * 256 + threadIdx.x) >> 5;
  int lane = threadIdx.x & 31;
  int mt = wave >> 3, nt = wave & 7;
  int row0 = mt * 16, col0 = nt * 16;
  v8f acc = {};
#pragma unroll
  for (int kk = 0; kk < 64; kk += 32) {
    v16h a = ld_a16(Xds, 64, row0, kk, lane);
    v16h b = ld_b16(Wh, 64, col0, kk, lane);
    acc = wmma16(a, b, acc);
  }
  int h = lane >> 4, nn = lane & 15, o = col0 + nn;
#pragma unroll
  for (int r = 0; r < 8; ++r) {
    int p = row0 + r + 8 * h;
    if (o < 125) ds_img[(size_t)p * 128 + o] = (_Float16)acc[r];
  }
}

// ---------------------------------------------------------------------------
// K5: bilinear upsample ds_img (quarter-res, channels-last f16) -> kv_ds
// ---------------------------------------------------------------------------
__global__ __launch_bounds__(256) void k_ds_up(const _Float16* ds_img,
                                               _Float16* kv_ds) {
  int p = blockIdx.x * 256 + threadIdx.x;
  int n = p >> 16, rem = p & 65535, y = rem >> 8, x = rem & 255;
  float sy = (y + 0.5f) * 0.5f - 0.5f, fy = floorf(sy), ty = sy - fy;
  float sx = (x + 0.5f) * 0.5f - 0.5f, fx = floorf(sx), tx = sx - fx;
  int y0 = min(max((int)fy, 0), 127), y1 = min(max((int)fy + 1, 0), 127);
  int x0 = min(max((int)fx, 0), 127), x1 = min(max((int)fx + 1, 0), 127);
  const _Float16* b00 = ds_img + ((size_t)(n * 16384 + y0 * 128 + x0)) * 128;
  const _Float16* b01 = ds_img + ((size_t)(n * 16384 + y0 * 128 + x1)) * 128;
  const _Float16* b10 = ds_img + ((size_t)(n * 16384 + y1 * 128 + x0)) * 128;
  const _Float16* b11 = ds_img + ((size_t)(n * 16384 + y1 * 128 + x1)) * 128;
  _Float16* dst = kv_ds + (size_t)p * 128;
#pragma unroll 4
  for (int c = 0; c < 128; ++c) {
    float top = (float)b00[c] + ((float)b01[c] - (float)b00[c]) * tx;
    float bot = (float)b10[c] + ((float)b11[c] - (float)b10[c]) * tx;
    dst[c] = (_Float16)(top + (bot - top) * ty);
  }
}

// ---------------------------------------------------------------------------
// K6: su conv1x1 GEMM via WMMA: su_emb = X @ su_w^T  (K=128, 4 k-steps)
// outputs 0..63  -> q tokens (window layout),
// outputs 64..188 -> kv_img f16 AND temporal_out f32 (channels 0..124)
// ---------------------------------------------------------------------------
__global__ __launch_bounds__(256) void k_gemm_su(const _Float16* X,
                                                 const _Float16* Wh,
                                                 _Float16* qws, _Float16* kv_img,
                                                 float* out_temporal) {
  int wave = (blockIdx.x * 256 + threadIdx.x) >> 5;
  int lane = threadIdx.x & 31;
  int mt = wave / 12, nt = wave % 12;
  int row0 = mt * 16, col0 = nt * 16;
  __builtin_prefetch(Wh + (size_t)col0 * 128, 0, 1);   // global_prefetch_b8
  v8f acc = {};
#pragma unroll
  for (int kk = 0; kk < 128; kk += 32) {
    v16h a = ld_a16(X, 128, row0, kk, lane);
    v16h b = ld_b16(Wh, 128, col0, kk, lane);
    acc = wmma16(a, b, acc);
  }
  int h = lane >> 4, nn = lane & 15, o = col0 + nn;
#pragma unroll
  for (int r = 0; r < 8; ++r) {
    int p = row0 + r + 8 * h;
    float v = acc[r];
    int n = p >> 16, rem = p & 65535;
    if (o < 64) {
      int y = rem >> 8, x = rem & 255;
      int win = n * 1024 + (y >> 3) * 32 + (x >> 3);
      int tok = (y & 7) * 8 + (x & 7);
      qws[((size_t)win * 64 + tok) * 64 + o] = (_Float16)v;
    } else if (o < 189) {
      int c2 = o - 64;
      kv_img[(size_t)p * 128 + c2] = (_Float16)v;
      out_temporal[((size_t)n * 128 + c2) * 65536 + rem] = v;
    }
  }
}

// ---------------------------------------------------------------------------
// K7: windowed attention, flash-style over 3 chunks of 128 KV tokens.
// Block = 128 threads (4 waves), one 8x8 window per block.
// LDS: Q 8K, K-tile 16K, V^T-tile 16K, P bounce 16K  (= 56 KB)
// ---------------------------------------------------------------------------
__global__ __launch_bounds__(128) void k_attn(const _Float16* qws,
                                              const _Float16* kv_img,
                                              const _Float16* kv_t,
                                              const _Float16* kv_ds,
                                              float* img_attn, float* d_out) {
  __shared__ _Float16 Qs[64 * 64];
  __shared__ _Float16 Ks[128 * 64];
  __shared__ _Float16 Vt[64 * 128];
  __shared__ _Float16 Ps[4][16 * 128];

  int tid = threadIdx.x, lane = tid & 31, w = tid >> 5;
  int win = blockIdx.x;
  int n = win >> 10, wloc = win & 1023, wy = wloc >> 5, wx = wloc & 31;

  { // load Q tile (64 tokens x 64 ch)
    const v8h* src = (const v8h*)(qws + (size_t)win * 4096);
    v8h* dst = (v8h*)Qs;
    for (int i = tid; i < 512; i += 128) dst[i] = src[i];
  }

  float mrun[8], lrun[8];
  v8f O[4];
#pragma unroll
  for (int r = 0; r < 8; ++r) { mrun[r] = -1e30f; lrun[r] = 0.f; }
#pragma unroll
  for (int ct = 0; ct < 4; ++ct) O[ct] = (v8f){};

  for (int chunk = 0; chunk < 3; ++chunk) {
    __syncthreads();           // previous PV gemm done before K/V overwrite
    { // stage 128 tokens: K row-major, V transposed
      int t = tid;
      const _Float16* src = nullptr;
      bool valid;
      if (chunk < 2) {         // kv_su: 16x16 band-4 window, zero-padded
        int s = chunk * 128 + t, sy = s >> 4, sx = s & 15;
        int gy = wy * 8 - 4 + sy, gx = wx * 8 - 4 + sx;
        valid = (gy >= 0) & (gy < 256) & (gx >= 0) & (gx < 256);
        if (valid) src = kv_img + ((size_t)(n * 65536 + gy * 256 + gx)) * 128;
      } else {                 // kv_t (64) then kv_ds (64)
        int tt = t & 63;
        int gy = wy * 8 + (tt >> 3), gx = wx * 8 + (tt & 7);
        src = ((t < 64) ? kv_t : kv_ds) +
              ((size_t)(n * 65536 + gy * 256 + gx)) * 128;
        valid = true;
      }
      if (valid) {
        const v8h* s8 = (const v8h*)src;
        v8h* k8 = (v8h*)(Ks + t * 64);
#pragma unroll
        for (int c8 = 0; c8 < 8; ++c8) k8[c8] = s8[c8];
#pragma unroll
        for (int c = 0; c < 64; ++c) Vt[c * 128 + t] = src[64 + c];
      } else {
        v8h z = (v8h){};
        v8h* k8 = (v8h*)(Ks + t * 64);
#pragma unroll
        for (int c8 = 0; c8 < 8; ++c8) k8[c8] = z;
#pragma unroll
        for (int c = 0; c < 64; ++c) Vt[c * 128 + t] = (_Float16)0.f;
      }
    }
    __syncthreads();

    // S stripe: 16 q-rows x 128 tokens, in registers (C layout)
    float Sv[8][8];
#pragma unroll
    for (int nt = 0; nt < 8; ++nt) {
      v8f s = {};
#pragma unroll
      for (int kk = 0; kk < 64; kk += 32) {
        v16h a = ld_a16(Qs, 64, w * 16, kk, lane);
        v16h b = ld_b16(Ks, 64, nt * 16, kk, lane);
        s = wmma16(a, b, s);
      }
#pragma unroll
      for (int r = 0; r < 8; ++r) Sv[nt][r] = s[r] * 0.125f;  // 1/sqrt(64)
    }

    // online softmax update (rows live across 16-lane halves)
    float mnew[8], alpha[8], rs[8];
#pragma unroll
    for (int r = 0; r < 8; ++r) {
      float mx = Sv[0][r];
#pragma unroll
      for (int nt = 1; nt < 8; ++nt) mx = fmaxf(mx, Sv[nt][r]);
      mx = redmax16(mx);
      mnew[r] = fmaxf(mrun[r], mx);
      alpha[r] = __expf(mrun[r] - mnew[r]);
      mrun[r] = mnew[r];
      rs[r] = 0.f;
    }
#pragma unroll
    for (int nt = 0; nt < 8; ++nt)
#pragma unroll
      for (int r = 0; r < 8; ++r) {
        float pv = __expf(Sv[nt][r] - mnew[r]);
        Sv[nt][r] = pv;
        rs[r] += pv;
      }
#pragma unroll
    for (int r = 0; r < 8; ++r) {
      lrun[r] = lrun[r] * alpha[r] + redsum16(rs[r]);
#pragma unroll
      for (int ct = 0; ct < 4; ++ct) O[ct][r] *= alpha[r];
    }

    // bounce P through LDS: C layout -> row-major A operand
    int h = lane >> 4, nn = lane & 15;
#pragma unroll
    for (int nt = 0; nt < 8; ++nt)
#pragma unroll
      for (int r = 0; r < 8; ++r)
        Ps[w][(r + 8 * h) * 128 + nt * 16 + nn] = (_Float16)Sv[nt][r];
    asm volatile("s_wait_dscnt 0" ::: "memory");   // same-wave LDS RAW fence

    // O += P @ V   (K=128 tokens, 4 k-steps, 4 output-channel tiles)
#pragma unroll
    for (int ct = 0; ct < 4; ++ct)
#pragma unroll
      for (int kk = 0; kk < 128; kk += 32) {
        v16h a = ld_a16(Ps[w], 128, 0, kk, lane);
        v16h b = ld_b16(Vt, 128, ct * 16, kk, lane);
        O[ct] = wmma16(a, b, O[ct]);
      }
  }

  // epilogue: normalize, scatter to img_attn; channels 61..63 also feed both outputs
  int h = lane >> 4, nn = lane & 15;
#pragma unroll
  for (int ct = 0; ct < 4; ++ct)
#pragma unroll
    for (int r = 0; r < 8; ++r) {
      int m = w * 16 + r + 8 * h;            // token in window
      float v = O[ct][r] / lrun[r];
      int gy = wy * 8 + (m >> 3), gx = wx * 8 + (m & 7);
      int rem = gy * 256 + gx;
      int p = n * 65536 + rem;
      int o = ct * 16 + nn;
      img_attn[(size_t)p * 64 + o] = v;
      if (o >= 61) {
        d_out[(size_t)DEC_OFF + ((size_t)n * 128 + 125 + (o - 61)) * 65536 + rem] = v;
        d_out[((size_t)n * 64 + o) * 65536 + rem] = v;   // decoded[61..63]
      }
    }
}

// ---------------------------------------------------------------------------
// K8: FFN: bn1 -> relu(W1 x + b1) -> bn2 -> W2 x + b2, two WMMA GEMMs.
// Block = 128 threads (4 waves) handles 64 pixels.
// ---------------------------------------------------------------------------
__global__ __launch_bounds__(128) void k_ffn(const float* img_attn,
                                             const _Float16* w1h, const float* b1,
                                             const float* bn1g, const float* bn1b,
                                             const float* bn1m, const float* bn1v,
                                             const float* bn2g, const float* bn2b,
                                             const float* bn2m, const float* bn2v,
                                             const _Float16* w2h, const float* b2,
                                             float* d_out) {
  __shared__ _Float16 A1[64 * 64];
  __shared__ _Float16 A2[64 * 128];
  int tid = threadIdx.x, lane = tid & 31, w = tid >> 5;
  size_t base = (size_t)blockIdx.x * 64;

  for (int e = tid; e < 4096; e += 128) {   // bn1 + f16
    int pl = e >> 6, c = e & 63;
    float xv = img_attn[(base + pl) * 64 + c];
    float inv = rsqrtf(bn1v[c] + EPSX);
    A1[pl * 64 + c] = (_Float16)((xv - bn1m[c]) * inv * bn1g[c] + bn1b[c]);
  }
  __syncthreads();

  int h = lane >> 4, nn = lane & 15;
#pragma unroll
  for (int nt = 0; nt < 8; ++nt) {          // GEMM1: 64 -> 128
    v8f acc = {};
#pragma unroll
    for (int kk = 0; kk < 64; kk += 32) {
      v16h a = ld_a16(A1, 64, w * 16, kk, lane);
      v16h b = ld_b16(w1h, 64, nt * 16, kk, lane);
      acc = wmma16(a, b, acc);
    }
    int o = nt * 16 + nn;
    float inv2 = rsqrtf(bn2v[o] + EPSX);
    float sc = inv2 * bn2g[o];
    float sh = bn2b[o] - bn2m[o] * sc;
    float bb = b1[o];
#pragma unroll
    for (int r = 0; r < 8; ++r) {
      float v = fmaxf(acc[r] + bb, 0.f);    // bias + relu
      A2[(w * 16 + r + 8 * h) * 128 + o] = (_Float16)(v * sc + sh);  // bn2
    }
  }
  __syncthreads();

#pragma unroll
  for (int ct = 0; ct < 4; ++ct) {          // GEMM2: 128 -> 61 (padded 64)
    v8f acc = {};
#pragma unroll
    for (int kk = 0; kk < 128; kk += 32) {
      v16h a = ld_a16(A2, 128, w * 16, kk, lane);
      v16h b = ld_b16(w2h, 128, ct * 16, kk, lane);
      acc = wmma16(a, b, acc);
    }
    int o = ct * 16 + nn;
    if (o < 61) {
      float bb = b2[o];
#pragma unroll
      for (int r = 0; r < 8; ++r) {
        size_t p = base + w * 16 + r + 8 * h;
        int n = (int)(p >> 16), rem = (int)(p & 65535);
        d_out[((size_t)n * 64 + o) * 65536 + rem] = acc[r] + bb;
      }
    }
  }
}

// ---------------------------------------------------------------------------
// Host launcher
// ---------------------------------------------------------------------------
extern "C" void kernel_launch(void* const* d_in, const int* in_sizes, int n_in,
                              void* d_out, int out_size, void* d_ws, size_t ws_size,
                              hipStream_t stream) {
  (void)in_sizes; (void)n_in; (void)out_size; (void)ws_size;
  const float* radiance    = (const float*)d_in[0];
  const float* skip_latent = (const float*)d_in[1];
  const float* ds_latent   = (const float*)d_in[2];
  const float* temporal_kv = (const float*)d_in[3];
  const float* motionvec   = (const float*)d_in[4];
  const float* su_ln_w     = (const float*)d_in[5];
  const float* su_ln_b     = (const float*)d_in[6];
  const float* su_w        = (const float*)d_in[7];
  const float* ds_ln_w     = (const float*)d_in[8];
  const float* ds_ln_b     = (const float*)d_in[9];
  const float* ds_w        = (const float*)d_in[10];
  const float* bn1_g       = (const float*)d_in[11];
  const float* bn1_b       = (const float*)d_in[12];
  const float* bn1_m       = (const float*)d_in[13];
  const float* bn1_v       = (const float*)d_in[14];
  const float* w1          = (const float*)d_in[15];
  const float* b1          = (const float*)d_in[16];
  const float* bn2_g       = (const float*)d_in[17];
  const float* bn2_b       = (const float*)d_in[18];
  const float* bn2_m       = (const float*)d_in[19];
  const float* bn2_v       = (const float*)d_in[20];
  const float* w2          = (const float*)d_in[21];
  const float* b2          = (const float*)d_in[22];
  float* out = (float*)d_out;

  char* ws = (char*)d_ws;
  size_t off = 0;
  auto carve = [&](size_t bytes) -> void* {
    void* p = ws + off;
    off += (bytes + 255) & ~(size_t)255;
    return p;
  };
  _Float16* X        = (_Float16*)carve((size_t)PIX * 128 * 2);
  _Float16* qws      = (_Float16*)carve((size_t)PIX * 64 * 2);
  _Float16* kv_img   = (_Float16*)carve((size_t)PIX * 128 * 2);
  _Float16* kv_t     = (_Float16*)carve((size_t)PIX * 128 * 2);
  _Float16* kv_ds    = (_Float16*)carve((size_t)PIX * 128 * 2);
  float*    img_attn = (float*)carve((size_t)PIX * 64 * 4);
  _Float16* Xds      = (_Float16*)carve((size_t)PIXD * 64 * 2);
  _Float16* ds_img   = (_Float16*)carve((size_t)PIXD * 128 * 2);
  _Float16* su_wh    = (_Float16*)carve(192 * 128 * 2);
  _Float16* ds_wh    = (_Float16*)carve(128 * 64 * 2);
  _Float16* w1h      = (_Float16*)carve(128 * 64 * 2);
  _Float16* w2h      = (_Float16*)carve(64 * 128 * 2);

  k_wconv<<<192, 256, 0, stream>>>(su_w, ds_w, w1, w2, su_wh, ds_wh, w1h, w2h);
  k_prep_su<<<PIX / 256, 256, 0, stream>>>(skip_latent, ds_latent, radiance,
                                           su_ln_w, su_ln_b, X, kv_img);
  k_warp<<<PIX / 256, 256, 0, stream>>>(temporal_kv, motionvec, kv_t);
  k_ds_ln<<<PIXD / 256, 256, 0, stream>>>(ds_latent, ds_ln_w, ds_ln_b, Xds, ds_img);
  k_gemm_ds<<<(PIXD / 16) * 8 / 8, 256, 0, stream>>>(Xds, ds_wh, ds_img);
  k_ds_up<<<PIX / 256, 256, 0, stream>>>(ds_img, kv_ds);
  k_gemm_su<<<(PIX / 16) * 12 / 8, 256, 0, stream>>>(X, su_wh, qws, kv_img,
                                                     out + DEC_OFF);
  k_attn<<<NB * 1024, 128, 0, stream>>>(qws, kv_img, kv_t, kv_ds, img_attn, out);
  k_ffn<<<PIX / 64, 128, 0, stream>>>(img_attn, w1h, b1, bn1_g, bn1_b, bn1_m,
                                      bn1_v, bn2_g, bn2_b, bn2_m, bn2_v,
                                      w2h, b2, out);
}
